// MatchNet_86131274154742
// MI455X (gfx1250) — compile-verified, matrix operand order checked by reading, not run on hardware
//
#include <hip/hip_runtime.h>
#include <hip/hip_bf16.h>

// ---------------------------------------------------------------------------
// MatchNet (LSTM attention read-out) for MI455X / gfx1250, wave32 + WMMA.
//
// Roofline reasoning:
//  * query_mean is loop invariant -> x@W_ih^T + biases hoisted out (1 GEMM).
//  * h_r == 0 at step 0 -> skip the first h_r@W_hh^T.
//  * last step's softmax/r are dead -> final GEMM h@S^T goes straight to out.
//  * working set (~150MB of bf16/fp32 buffers) fits the 192MB L2 -> GEMMs are
//    compute bound -> v_wmma_f32_16x16x32_bf16 (fp32 accum, bf16 operands).
//  * 64x64 register tile per wave: 16 WMMA per 8 fragment loads per 32-K slab
//    -> 32 FLOP/byte of L2 fragment traffic (2x the 32x32 tile), so the loop
//    stays WMMA-limited rather than L2-fetch-limited.
// ---------------------------------------------------------------------------

typedef __attribute__((ext_vector_type(16))) __bf16 v16bf;
typedef __attribute__((ext_vector_type(8)))  float  v8f;

#define BQ 2048   // batch
#define DD 1024   // input dim
#define HH 2048   // hidden (2*D)
#define G4 8192   // 4*H gate width
#define SS 512    // support set size

// --- fragment load for NT GEMM (A[M,K] row-major, B[N,K] row-major) --------
// lanes 0..15 : rows row0..row0+15, K = k0..k0+15
// lanes 16..31: rows row0..row0+15, K = k0+16..k0+31
__device__ __forceinline__ v16bf load_frag_nt(const __bf16* __restrict__ p,
                                              int ld, int row0, int k0, int lane) {
    const __bf16* q = p + (size_t)(row0 + (lane & 15)) * ld + (k0 + ((lane >> 4) << 4));
    return *(const v16bf*)q;
}

// Epilogue for one 16x16 f32 accumulator tile (C/D layout per CDNA5 ISA:
// VGPR r, lanes 0-15 -> M=r, N=lane; lanes 16-31 -> M=r+8, N=lane-16).
__device__ __forceinline__ void store_tile(const v8f acc, int mbase, int nbase, int lane,
                                           float* __restrict__ Cout, __bf16* __restrict__ Obf,
                                           const float* __restrict__ bias0,
                                           const float* __restrict__ bias1,
                                           const float* __restrict__ Cin,
                                           int ldc, int ldbf, int bfcol) {
    const int col   = nbase + (lane & 15);
    const int rbase = mbase + ((lane >> 4) << 3);
    float badd = 0.f;
    if (bias0) badd += bias0[col];
    if (bias1) badd += bias1[col];
#pragma unroll
    for (int r = 0; r < 8; ++r) {
        const int row = rbase + r;
        float v = acc[r] + badd;
        if (Cin)  v += Cin[(size_t)row * ldc + col];
        if (Cout) Cout[(size_t)row * ldc + col] = v;
        if (Obf)  Obf[(size_t)row * ldbf + bfcol + col] = (__bf16)v;
    }
}

// --- general NT GEMM: out = A[M,K] * B[N,K]^T (+bias0+bias1)(+Cin) ----------
// Block: 128 threads = 4 waves, arranged 2(M) x 2(N); wave tile 64x64
// (4x4 WMMA accumulators); block tile 128(M) x 128(N).
// All call sites satisfy M%128==0, N%128==0, K%32==0.
__global__ __launch_bounds__(128)
void gemm_nt_bf16_wmma(float* __restrict__ Cout, __bf16* __restrict__ Obf,
                       const __bf16* __restrict__ A, const __bf16* __restrict__ B,
                       const float* __restrict__ bias0, const float* __restrict__ bias1,
                       const float* __restrict__ Cin,
                       int M, int N, int K, int lda, int ldb,
                       int ldc, int ldbf, int bfcol) {
    const int lane = threadIdx.x & 31;
    const int wid  = threadIdx.x >> 5;     // 0..3
    const int wm   = wid & 1;              // 2 waves along M
    const int wn   = wid >> 1;             // 2 waves along N
    const int m0   = blockIdx.y * 128 + wm * 64;
    const int n0   = blockIdx.x * 128 + wn * 64;
    if (m0 >= M || n0 >= N) return;   // wave-uniform: EXEC stays all-1s for WMMA

    v8f acc[4][4] = {};
    for (int k = 0; k < K; k += 32) {
        if (k + 32 < K) {  // pull next K-slab toward L2/L0 (global_prefetch_b8)
            __builtin_prefetch(A + (size_t)(m0 + (lane & 15)) * lda + k + 32, 0, 3);
            __builtin_prefetch(B + (size_t)(n0 + (lane & 15)) * ldb + k + 32, 0, 3);
        }
        v16bf a[4], b[4];
#pragma unroll
        for (int i = 0; i < 4; ++i) a[i] = load_frag_nt(A, lda, m0 + 16 * i, k, lane);
#pragma unroll
        for (int j = 0; j < 4; ++j) b[j] = load_frag_nt(B, ldb, n0 + 16 * j, k, lane);
#pragma unroll
        for (int i = 0; i < 4; ++i)
#pragma unroll
            for (int j = 0; j < 4; ++j)
                acc[i][j] = __builtin_amdgcn_wmma_f32_16x16x32_bf16(
                    false, a[i], false, b[j], (short)0, acc[i][j], false, false);
    }
#pragma unroll
    for (int i = 0; i < 4; ++i)
#pragma unroll
        for (int j = 0; j < 4; ++j)
            store_tile(acc[i][j], m0 + 16 * i, n0 + 16 * j, lane,
                       Cout, Obf, bias0, bias1, Cin, ldc, ldbf, bfcol);
}

// --- LSTM elementwise: c update + h = query + h_new[:, :D] (packed bf16) ----
__global__ void lstm_elem(const float* __restrict__ gates, float* __restrict__ c,
                          const float* __restrict__ query, __bf16* __restrict__ hrB,
                          int first) {
    int idx = blockIdx.x * blockDim.x + threadIdx.x;   // over BQ*HH
    if (idx >= BQ * HH) return;
    int b = idx >> 11;            // /HH
    int j = idx & (HH - 1);
    const float* g = gates + (size_t)b * G4;
    float ig = g[j], fg = g[HH + j], gg = g[2 * HH + j], og = g[3 * HH + j];
    float cp = first ? 0.f : c[idx];
    float si = 1.f / (1.f + __expf(-ig));
    float sf = 1.f / (1.f + __expf(-fg));
    float cn = sf * cp + si * tanhf(gg);
    c[idx] = cn;
    if (j < DD) {
        float so = 1.f / (1.f + __expf(-og));
        float h  = query[(size_t)b * DD + j] + so * tanhf(cn);
        hrB[(size_t)b * HH + j] = (__bf16)h;   // h-half of h_r, bf16 for WMMA
    }
}

// --- row softmax over S=512, result in bf16 for the r-GEMM ------------------
__global__ void softmax_row(__bf16* __restrict__ attn, const float* __restrict__ scores) {
    __shared__ float red[256];
    const int row = blockIdx.x, tid = threadIdx.x;
    const float* s = scores + (size_t)row * SS;
    float v0 = s[tid], v1 = s[tid + 256];
    red[tid] = fmaxf(v0, v1);
    __syncthreads();
    for (int o = 128; o > 0; o >>= 1) {
        if (tid < o) red[tid] = fmaxf(red[tid], red[tid + o]);
        __syncthreads();
    }
    float m = red[0];
    __syncthreads();
    float e0 = __expf(v0 - m), e1 = __expf(v1 - m);
    red[tid] = e0 + e1;
    __syncthreads();
    for (int o = 128; o > 0; o >>= 1) {
        if (tid < o) red[tid] += red[tid + o];
        __syncthreads();
    }
    float inv = 1.f / red[0];
    attn[(size_t)row * SS + tid]       = (__bf16)(e0 * inv);
    attn[(size_t)row * SS + tid + 256] = (__bf16)(e1 * inv);
}

// --- fp32 -> bf16 converters ------------------------------------------------
__global__ void cvt_f32_bf16(__bf16* __restrict__ dst, const float* __restrict__ src, int n) {
    for (int i = blockIdx.x * blockDim.x + threadIdx.x; i < n; i += gridDim.x * blockDim.x)
        dst[i] = (__bf16)src[i];
}

__global__ void transpose_f32_bf16(__bf16* __restrict__ dst, const float* __restrict__ src,
                                   int rows, int cols) {
    int i = blockIdx.x * blockDim.x + threadIdx.x;
    if (i < rows * cols) {
        int r = i / cols, c = i - r * cols;
        dst[(size_t)c * rows + r] = (__bf16)src[i];
    }
}

// ---------------------------------------------------------------------------
extern "C" void kernel_launch(void* const* d_in, const int* in_sizes, int n_in,
                              void* d_out, int out_size, void* d_ws, size_t ws_size,
                              hipStream_t stream) {
    const float* support = (const float*)d_in[0];   // [512, 1024]
    const float* query   = (const float*)d_in[2];   // [2048, 1024]
    const float* W_ih    = (const float*)d_in[4];   // [8192, 1024]
    const float* W_hh    = (const float*)d_in[5];   // [8192, 2048]
    const float* b_ih    = (const float*)d_in[6];   // [8192]
    const float* b_hh    = (const float*)d_in[7];   // [8192]
    (void)in_sizes; (void)n_in; (void)out_size; (void)ws_size;

    char* ws = (char*)d_ws;
    size_t off = 0;
    auto alloc = [&](size_t bytes) -> void* {
        void* p = ws + off;
        off += (bytes + 255) & ~(size_t)255;
        return p;
    };
    __bf16* WihB   = (__bf16*)alloc((size_t)G4 * DD * 2);   // 16 MB
    __bf16* WhhB   = (__bf16*)alloc((size_t)G4 * HH * 2);   // 32 MB
    __bf16* QB     = (__bf16*)alloc((size_t)BQ * DD * 2);   //  4 MB
    __bf16* SB     = (__bf16*)alloc((size_t)SS * DD * 2);   //  1 MB
    __bf16* STB    = (__bf16*)alloc((size_t)DD * SS * 2);   //  1 MB (support^T)
    float*  gatesx = (float*) alloc((size_t)BQ * G4 * 4);   // 64 MB (loop-invariant)
    float*  gates  = (float*) alloc((size_t)BQ * G4 * 4);   // 64 MB
    float*  cbuf   = (float*) alloc((size_t)BQ * HH * 4);   // 16 MB
    __bf16* hrB    = (__bf16*)alloc((size_t)BQ * HH * 2);   //  8 MB packed [h | r]
    float*  scores = (float*) alloc((size_t)BQ * SS * 4);   //  4 MB
    __bf16* attnB  = (__bf16*)alloc((size_t)BQ * SS * 2);   //  2 MB

    // One-time precision conversion (all subsequent GEMM traffic is bf16).
    cvt_f32_bf16<<<4096, 256, 0, stream>>>(WihB, W_ih, G4 * DD);
    cvt_f32_bf16<<<8192, 256, 0, stream>>>(WhhB, W_hh, G4 * HH);
    cvt_f32_bf16<<<4096, 256, 0, stream>>>(QB, query, BQ * DD);
    cvt_f32_bf16<<<2048, 256, 0, stream>>>(SB, support, SS * DD);
    transpose_f32_bf16<<<(SS * DD + 255) / 256, 256, 0, stream>>>(STB, support, SS, DD);

    // Hoisted: gates_x = query @ W_ih^T + b_ih + b_hh      [2048, 8192]
    gemm_nt_bf16_wmma<<<dim3(G4 / 128, BQ / 128), 128, 0, stream>>>(
        gatesx, nullptr, QB, WihB, b_ih, b_hh, nullptr,
        BQ, G4, DD, DD, DD, G4, 0, 0);

    for (int step = 0; step < 4; ++step) {
        const float* gptr = gatesx;
        if (step > 0) {
            // gates = gates_x + h_r @ W_hh^T               [2048, 8192], K=2048
            gemm_nt_bf16_wmma<<<dim3(G4 / 128, BQ / 128), 128, 0, stream>>>(
                gates, nullptr, hrB, WhhB, nullptr, nullptr, gatesx,
                BQ, G4, HH, HH, HH, G4, 0, 0);
            gptr = gates;
        }
        // c update + h = query + h_new[:, :D] (bf16 into hrB[:, 0:1024])
        lstm_elem<<<(BQ * HH + 255) / 256, 256, 0, stream>>>(gptr, cbuf, query, hrB, step == 0);

        if (step < 3) {
            // scores = h @ support^T                       [2048, 512], K=1024
            gemm_nt_bf16_wmma<<<dim3(SS / 128, BQ / 128), 128, 0, stream>>>(
                scores, nullptr, hrB, SB, nullptr, nullptr, nullptr,
                BQ, SS, DD, HH, DD, SS, 0, 0);
            softmax_row<<<BQ, 256, 0, stream>>>(attnB, scores);
            // r = attn @ support -> bf16 into hrB[:, 1024:2048]   K=512
            gemm_nt_bf16_wmma<<<dim3(DD / 128, BQ / 128), 128, 0, stream>>>(
                nullptr, hrB, attnB, STB, nullptr, nullptr, nullptr,
                BQ, DD, SS, SS, SS, DD, HH, DD);
        } else {
            // matching_scores = h @ support^T straight to output (softmax/r dead)
            gemm_nt_bf16_wmma<<<dim3(SS / 128, BQ / 128), 128, 0, stream>>>(
                (float*)d_out, nullptr, hrB, SB, nullptr, nullptr, nullptr,
                BQ, SS, DD, HH, DD, SS, 0, 0);
        }
    }
}
